// SparseLinearCrossAttention_2877628088582
// MI455X (gfx1250) — matrix operand order repaired and checked by reading, not compiled, not run on hardware
//
#include <hip/hip_runtime.h>
#include <hip/hip_bf16.h>

// ---------------- types / helpers ----------------
typedef __attribute__((ext_vector_type(16))) __bf16 v16bf;
typedef __attribute__((ext_vector_type(8)))  float  v8f;

union BFrag { v16bf bf; unsigned short u[16]; };

__device__ __forceinline__ unsigned short f2bf(float x) {
  union { float f; unsigned int u; } c; c.f = x;
  unsigned int r = c.u + 0x7FFFu + ((c.u >> 16) & 1u);   // round-to-nearest-even
  return (unsigned short)(r >> 16);
}

#define NHEAD 8
#define SEQ   4096
#define DIM   64
#define NBLK  64      // SEQ / 64
#define TOPK  16

// ---- CDNA5 async global->LDS copy (signature per hipcc diagnostic) ----
typedef int v4i_ __attribute__((vector_size(4 * sizeof(int))));
typedef __attribute__((address_space(1))) v4i_* g_v4i_p;   // global (AS1)
typedef __attribute__((address_space(3))) v4i_* l_v4i_p;   // LDS (AS3)

__device__ __forceinline__ void async_copy_b128(const float* g, float* l) {
#if __has_builtin(__builtin_amdgcn_global_load_async_to_lds_b128)
  __builtin_amdgcn_global_load_async_to_lds_b128((g_v4i_p)g, (l_v4i_p)l, 0, 0);
#else
  *(float4*)l = *(const float4*)g;                       // sync fallback
#endif
}

__device__ __forceinline__ void wait_async0() {
#if __has_builtin(__builtin_amdgcn_s_wait_asynccnt)
  __builtin_amdgcn_s_wait_asynccnt(0);
#else
  asm volatile("s_wait_asynccnt 0" ::: "memory");
#endif
}

// ---------------- 1) block mean pooling (pq, pk) ----------------
__global__ __launch_bounds__(64) void pool_kernel(const float* __restrict__ q,
                                                  const float* __restrict__ k,
                                                  float* __restrict__ pq,
                                                  float* __restrict__ pk) {
  int blk = blockIdx.x, h = blockIdx.y, d = threadIdx.x;
  const float* qb = q + ((size_t)(h * SEQ) + blk * 64) * DIM;
  const float* kb = k + ((size_t)(h * SEQ) + blk * 64) * DIM;
  float sq = 0.f, sk = 0.f;
  for (int r = 0; r < 64; ++r) { sq += qb[r * DIM + d]; sk += kb[r * DIM + d]; }
  pq[(h * NBLK + blk) * DIM + d] = sq * (1.f / 64.f);
  pk[(h * NBLK + blk) * DIM + d] = sk * (1.f / 64.f);
}

// ---------------- 2) pooled score + top-16 routing ----------------
__global__ __launch_bounds__(64) void topk_kernel(const float* __restrict__ pq,
                                                  const float* __restrict__ pk,
                                                  int* __restrict__ lut) {
  int qb = blockIdx.x, h = blockIdx.y, t = threadIdx.x;   // t = candidate k-block
  __shared__ float sc[NBLK];
  __shared__ float qv[DIM];
  qv[t] = pq[(h * NBLK + qb) * DIM + t];
  __syncthreads();
  float a = 0.f;
  const float* pkr = pk + (h * NBLK + t) * DIM;
  for (int d = 0; d < DIM; ++d) a += qv[d] * pkr[d];
  sc[t] = a;
  __syncthreads();
  if (t == 0) {
    for (int s = 0; s < TOPK; ++s) {
      int bi = 0; float bv = -1e30f;
      for (int j = 0; j < NBLK; ++j) if (sc[j] > bv) { bv = sc[j]; bi = j; }
      lut[((h * NBLK) + qb) * TOPK + s] = bi;
      sc[bi] = -1e31f;
    }
  }
}

// ---------------- 3) linear-branch stats: kvp = (kf^T v) @ W^T, kfsum ----------------
__global__ __launch_bounds__(256) void linstat_kernel(const float* __restrict__ k,
                                                      const float* __restrict__ v,
                                                      const float* __restrict__ proj_w,
                                                      float* __restrict__ kvp_out,
                                                      float* __restrict__ kfs_out) {
  int h = blockIdx.x, tid = threadIdx.x;
  __shared__ float KF[64 * 64];   // softmax(k) chunk, later reused for kv
  __shared__ float VS[64 * 64];   // v chunk
  int dd = tid >> 2;              // 0..63 : feature d
  int e0 = (tid & 3) * 16;        // 16-wide e strip
  float acc[16];
#pragma unroll
  for (int j = 0; j < 16; ++j) acc[j] = 0.f;
  float ksum = 0.f;

  for (int ch = 0; ch < 64; ++ch) {                 // 64 chunks x 64 rows
    __syncthreads();
    if (tid < 64) {
      const float* kr = k + ((size_t)(h * SEQ) + ch * 64 + tid) * DIM;
      float tmp[DIM]; float m = -1e30f;
#pragma unroll
      for (int d = 0; d < DIM; ++d) { tmp[d] = kr[d]; m = fmaxf(m, tmp[d]); }
      float s = 0.f;
#pragma unroll
      for (int d = 0; d < DIM; ++d) { tmp[d] = __expf(tmp[d] - m); s += tmp[d]; }
      float inv = 1.f / s;
#pragma unroll
      for (int d = 0; d < DIM; ++d) KF[tid * DIM + d] = tmp[d] * inv;
    }
    const float* vc = v + ((size_t)(h * SEQ) + ch * 64) * DIM;
    for (int i = tid; i < 64 * DIM; i += 256) VS[i] = vc[i];
    __syncthreads();
    for (int rr = 0; rr < 64; ++rr) {
      float kfv = KF[rr * DIM + dd];
      ksum += kfv;
#pragma unroll
      for (int j = 0; j < 16; ++j) acc[j] += kfv * VS[rr * DIM + e0 + j];
    }
  }
  __syncthreads();
#pragma unroll
  for (int j = 0; j < 16; ++j) KF[dd * DIM + e0 + j] = acc[j];   // KF now holds kv
  __syncthreads();
  // kvp[d][e] = sum_d' kv[d][d'] * W[e][d']   (fold projection; o_l@W^T)
#pragma unroll
  for (int j = 0; j < 16; ++j) {
    int e = e0 + j;
    float a = 0.f;
    const float* wr = proj_w + e * DIM;
    for (int dp = 0; dp < DIM; ++dp) a += KF[dd * DIM + dp] * wr[dp];
    kvp_out[h * DIM * DIM + dd * DIM + e] = a;
  }
  if ((tid & 3) == 0) kfs_out[h * DIM + dd] = ksum;
}

// ---------------- 4) fused block-sparse attention + linear branch ----------------
__global__ __launch_bounds__(128) void attn_kernel(const float* __restrict__ q,
                                                   const float* __restrict__ k,
                                                   const float* __restrict__ v,
                                                   const float* __restrict__ kvp,
                                                   const float* __restrict__ kfsum,
                                                   const float* __restrict__ proj_b,
                                                   const int* __restrict__ lut,
                                                   float* __restrict__ out) {
  const int qblk = blockIdx.x;          // 0..63
  const int h    = blockIdx.y;          // 0..7
  const int tid  = threadIdx.x;         // 0..127 (4 waves)
  const int lane = tid & 31;
  const int wave = tid >> 5;

  __shared__ unsigned short Qbf[64 * DIM];          // 8 KB
  __shared__ unsigned short Pst[4][16 * DIM];       // 8 KB, wave-private strips
  __shared__ float Kvp[DIM * DIM];                  // 16 KB
  __shared__ float Kfs[DIM];
  __shared__ float Bias[DIM];
  __shared__ float KV[2][2 * 64 * DIM];             // 64 KB double buffer: [buf][K|V] f32

  const size_t rowbase = (size_t)(h * SEQ) + qblk * 64;
  const float* qh = q + rowbase * DIM;
  const int lutbase = ((h * NBLK) + qblk) * TOPK;

  // ---- issue async copy of the first K/V block into KV[0] ----
  {
    const int kb0 = lut[lutbase];
    const float* kp = k + ((size_t)(h * SEQ) + kb0 * 64) * DIM;
    const float* vp = v + ((size_t)(h * SEQ) + kb0 * 64) * DIM;
    for (int i = tid; i < 1024; i += 128) {           // 1024 x float4 per matrix
      async_copy_b128(kp + i * 4, &KV[0][i * 4]);
      async_copy_b128(vp + i * 4, &KV[0][4096 + i * 4]);
    }
  }

  // ---- stage Q (bf16), kvp, kfsum, bias ----
  for (int i = tid; i < 64 * DIM; i += 128) Qbf[i] = f2bf(qh[i]);
  for (int i = tid; i < DIM * DIM; i += 128) Kvp[i] = kvp[h * DIM * DIM + i];
  if (tid < DIM) { Kfs[tid] = kfsum[h * DIM + tid]; Bias[tid] = proj_b[tid]; }
  wait_async0();
  __syncthreads();

  // ---- load Q strip as two A-fragments (16x32 bf16 each) ----
  const int am  = lane & 15;            // A: lane -> row M
  const int akb = (lane >> 4) * 16;     // A: K base per lane half
  const int m0  = wave * 16;
  BFrag qa[2];
#pragma unroll
  for (int c = 0; c < 2; ++c)
#pragma unroll
    for (int i = 0; i < 16; ++i) {
      int d = c * 32 + akb + i;
      qa[c].u[i] = Qbf[(m0 + am) * DIM + d];
    }

  v8f acc[4];
#pragma unroll
  for (int c = 0; c < 4; ++c) acc[c] = (v8f){0.f,0.f,0.f,0.f,0.f,0.f,0.f,0.f};
  float rmax[8], rsum[8];
#pragma unroll
  for (int r = 0; r < 8; ++r) { rmax[r] = -1e30f; rsum[r] = 0.f; }

  const int bn0 = lane & 15;            // B: lane -> col N (within tile)
  const int bk0 = (lane >> 4) * 16;     // B: K base per lane half

  for (int t = 0; t < TOPK; ++t) {
    const int cur = t & 1;
    // ---- prefetch next block into the other buffer (overlaps with compute) ----
    if (t + 1 < TOPK) {
      const int kbn = lut[lutbase + t + 1];
      const float* kp = k + ((size_t)(h * SEQ) + kbn * 64) * DIM;
      const float* vp = v + ((size_t)(h * SEQ) + kbn * 64) * DIM;
      float* dst = KV[1 - cur];
      for (int i = tid; i < 1024; i += 128) {
        async_copy_b128(kp + i * 4, dst + i * 4);
        async_copy_b128(vp + i * 4, dst + 4096 + i * 4);
      }
    }
    const float* Kc = &KV[cur][0];
    const float* Vc = &KV[cur][4096];

    // ---- S = Q . K^T  (4 col tiles of 16, K-dim = d = 64 -> 2 chained WMMAs) ----
    v8f s[4];
#pragma unroll
    for (int c = 0; c < 4; ++c) {
      BFrag b0, b1;
      int n = c * 16 + bn0;
#pragma unroll
      for (int i = 0; i < 16; ++i) {
        b0.u[i] = f2bf(Kc[n * DIM + bk0 + i]);        // d = 0..31
        b1.u[i] = f2bf(Kc[n * DIM + 32 + bk0 + i]);   // d = 32..63
      }
      v8f z = (v8f){0.f,0.f,0.f,0.f,0.f,0.f,0.f,0.f};
      v8f t0 = __builtin_amdgcn_wmma_f32_16x16x32_bf16(false, qa[0].bf, false, b0.bf,
                                                       (short)0, z, false, false);
      s[c] = __builtin_amdgcn_wmma_f32_16x16x32_bf16(false, qa[1].bf, false, b1.bf,
                                                     (short)0, t0, false, false);
      s[c] *= 0.125f;                                // D^-0.5
    }

    // ---- online softmax (rows live in 16-lane halves) ----
    float nm[8];
#pragma unroll
    for (int r = 0; r < 8; ++r) {
      float m = fmaxf(fmaxf(s[0][r], s[1][r]), fmaxf(s[2][r], s[3][r]));
#pragma unroll
      for (int off = 1; off < 16; off <<= 1) m = fmaxf(m, __shfl_xor(m, off, 32));
      nm[r] = fmaxf(rmax[r], m);
    }
#pragma unroll
    for (int r = 0; r < 8; ++r) {
      float alpha = __expf(rmax[r] - nm[r]);
      rmax[r] = nm[r];
      rsum[r] *= alpha;
#pragma unroll
      for (int c = 0; c < 4; ++c) acc[c][r] *= alpha;
    }
#pragma unroll
    for (int r = 0; r < 8; ++r) {
      float ps = 0.f;
#pragma unroll
      for (int c = 0; c < 4; ++c) {
        float p = __expf(s[c][r] - rmax[r]);
        s[c][r] = p; ps += p;
      }
#pragma unroll
      for (int off = 1; off < 16; off <<= 1) ps += __shfl_xor(ps, off, 32);
      rsum[r] += ps;
    }

    // ---- relayout P: C-layout -> LDS -> A-layout (wave-private) ----
#pragma unroll
    for (int c = 0; c < 4; ++c)
#pragma unroll
      for (int r = 0; r < 8; ++r)
        Pst[wave][((lane >> 4) * 8 + r) * DIM + c * 16 + (lane & 15)] = f2bf(s[c][r]);
    asm volatile("s_wait_dscnt 0" ::: "memory");     // same-wave LDS RAW

    BFrag pa[2];
#pragma unroll
    for (int c = 0; c < 2; ++c)
#pragma unroll
      for (int i = 0; i < 16; ++i) {
        int kk = c * 32 + akb + i;   // key index 0..63
        pa[c].u[i] = Pst[wave][am * DIM + kk];
      }

    // ---- O += P . V ----
#pragma unroll
    for (int c = 0; c < 4; ++c) {
      BFrag b0, b1;
      int n = c * 16 + bn0;                          // d column
#pragma unroll
      for (int i = 0; i < 16; ++i) {
        b0.u[i] = f2bf(Vc[(bk0 + i) * DIM + n]);      // keys 0..31
        b1.u[i] = f2bf(Vc[(32 + bk0 + i) * DIM + n]); // keys 32..63
      }
      acc[c] = __builtin_amdgcn_wmma_f32_16x16x32_bf16(false, pa[0].bf, false, b0.bf,
                                                       (short)0, acc[c], false, false);
      acc[c] = __builtin_amdgcn_wmma_f32_16x16x32_bf16(false, pa[1].bf, false, b1.bf,
                                                       (short)0, acc[c], false, false);
    }

    // ---- retire this buffer; prefetch for t+1 must be complete before reuse ----
    wait_async0();
    __syncthreads();
  }

  // ---- normalized o_s into LDS (reuse KV[0]) ----
  float* Osm = &KV[0][0];
#pragma unroll
  for (int c = 0; c < 4; ++c)
#pragma unroll
    for (int r = 0; r < 8; ++r) {
      int m = wave * 16 + (lane >> 4) * 8 + r;
      int n = c * 16 + (lane & 15);
      Osm[m * DIM + n] = acc[c][r] / rsum[r];
    }
  __syncthreads();

  // ---- linear branch + write out: 2 threads per row, 32 e's each ----
  {
    int row = tid >> 1;
    int e0  = (tid & 1) * 32;
    const float* qrow = qh + row * DIM;
    float qf[DIM]; float mx = -1e30f;
#pragma unroll
    for (int d = 0; d < DIM; ++d) { qf[d] = qrow[d]; mx = fmaxf(mx, qf[d]); }
    float se = 0.f;
#pragma unroll
    for (int d = 0; d < DIM; ++d) { qf[d] = __expf(qf[d] - mx); se += qf[d]; }
    float inv = 1.f / se;
    float denom = 0.f;
#pragma unroll
    for (int d = 0; d < DIM; ++d) { qf[d] *= inv; denom += qf[d] * Kfs[d]; }
    float rdenom = 1.f / (denom + 1e-5f);
    float* orow = out + (rowbase + row) * DIM;
    for (int e = e0; e < e0 + 32; ++e) {
      float a = 0.f;
#pragma unroll
      for (int d = 0; d < DIM; ++d) a += qf[d] * Kvp[d * DIM + e];
      orow[e] = Osm[row * DIM + e] + a * rdenom + Bias[e];
    }
  }
}

// ---------------- launcher ----------------
extern "C" void kernel_launch(void* const* d_in, const int* in_sizes, int n_in,
                              void* d_out, int out_size, void* d_ws, size_t ws_size,
                              hipStream_t stream) {
  const float* q  = (const float*)d_in[0];
  const float* k  = (const float*)d_in[1];
  const float* v  = (const float*)d_in[2];
  const float* pw = (const float*)d_in[3];
  const float* pb = (const float*)d_in[4];
  float* out = (float*)d_out;

  // workspace carve-up (~430 KB)
  char* ws = (char*)d_ws;
  float* pq  = (float*)ws;                         // 8*64*64
  float* pk  = pq + NHEAD * NBLK * DIM;            // 8*64*64
  int*   lut = (int*)(pk + NHEAD * NBLK * DIM);    // 8*64*16
  float* kvp = (float*)(lut + NHEAD * NBLK * TOPK);// 8*64*64
  float* kfs = kvp + NHEAD * DIM * DIM;            // 8*64

  pool_kernel<<<dim3(NBLK, NHEAD), 64, 0, stream>>>(q, k, pq, pk);
  topk_kernel<<<dim3(NBLK, NHEAD), 64, 0, stream>>>(pq, pk, lut);
  linstat_kernel<<<dim3(NHEAD), 256, 0, stream>>>(k, v, pw, kvp, kfs);
  attn_kernel<<<dim3(NBLK, NHEAD), 128, 0, stream>>>(q, k, v, kvp, kfs, pb, lut, out);
}
